// SceneFlowPWC_5291399708677
// MI455X (gfx1250) — matrix operand order repaired and checked by reading, not compiled, not run on hardware
//
#include <hip/hip_runtime.h>

typedef __attribute__((ext_vector_type(16))) _Float16 v16h;
typedef __attribute__((ext_vector_type(8)))  float    v8f;

#define B_ 4
#define N_ 8192
#define M_ 8192
#define D_ 64
#define K_ 16
#define LEAKY_ 0.1f

__device__ __forceinline__ float leaky(float x) {
  return fmaxf(x, LEAKY_ * x);   // valid for 0 < slope < 1
}

// ---------------------------------------------------------------------------
// Kernel 1: brute-force KNN (top-16 smallest squared distances per query).
// One thread = one query point. pc2 tiled through LDS (shared across block),
// register-resident insertion-sorted top-16.
// ---------------------------------------------------------------------------
#define KNN_TILE 256

__global__ __launch_bounds__(256) void knn_kernel(const float* __restrict__ pc1,
                                                  const float* __restrict__ pc2,
                                                  int* __restrict__ idx_out) {
  __shared__ float sx[KNN_TILE], sy[KNN_TILE], sz[KNN_TILE];
  const int tid = threadIdx.x;
  const int gq  = blockIdx.x * 256 + tid;   // global query id in [0, B*N)
  const int b   = gq / N_;

  const float qx = pc1[(size_t)gq * 3 + 0];
  const float qy = pc1[(size_t)gq * 3 + 1];
  const float qz = pc1[(size_t)gq * 3 + 2];

  float bestd[K_];
  int   besti[K_];
#pragma unroll
  for (int i = 0; i < K_; ++i) { bestd[i] = 3.4e38f; besti[i] = 0; }

  const float* p2 = pc2 + (size_t)b * M_ * 3;

  for (int m0 = 0; m0 < M_; m0 += KNN_TILE) {
    sx[tid] = p2[(m0 + tid) * 3 + 0];
    sy[tid] = p2[(m0 + tid) * 3 + 1];
    sz[tid] = p2[(m0 + tid) * 3 + 2];
    __syncthreads();
    if (m0 + KNN_TILE < M_)
      __builtin_prefetch(&p2[(m0 + KNN_TILE + tid) * 3], 0, 1);  // global_prefetch_b8

    for (int j = 0; j < KNN_TILE; ++j) {
      const float dx = sx[j] - qx;
      const float dy = sy[j] - qy;
      const float dz = sz[j] - qz;
      float d = dx * dx + dy * dy + dz * dz;
      if (d < bestd[K_ - 1]) {           // rare path (~1.2% of candidates)
        float dd = d; int ii = m0 + j;
#pragma unroll
        for (int p = 0; p < K_; ++p) {
          if (dd < bestd[p]) {
            const float td = bestd[p]; const int ti = besti[p];
            bestd[p] = dd; besti[p] = ii;
            dd = td; ii = ti;
          }
        }
      }
    }
    __syncthreads();
  }

#pragma unroll
  for (int kk = 0; kk < K_; ++kk)
    idx_out[(size_t)gq * K_ + kk] = besti[kk];
}

// ---------------------------------------------------------------------------
// Kernel 2: gather + [67->64] MLP + LeakyReLU + [64->64] MLP + LeakyReLU
//           + max-pool over 16 neighbors.
//
// One wave32 handles QPW=2 query points; the 16 neighbors of a query are the
// 16 rows of a WMMA A-tile, so max-pool = per-lane max over 8 C VGPRs + one
// shfl_xor(16).
//
// Input channels are permuted to [feat(0..63) | bias=1 | xyz | zero-pad]
// (W1 rows permuted identically), so:
//   * k-steps 0/1 of A are aligned float4 gathers (global_load_b128)
//   * k-step 2 is {1,nx,ny,nz,0...} for half the lanes, all-zero for the rest
//   * layer-1 bias rides the GEMM; accumulators start as inline-0 SRC2
//   * weights pre-packed in B-register layout -> one v16h (2x ds_load_b128)
//   * layer-1 output stored directly in A-register layout -> contiguous reads
// ---------------------------------------------------------------------------
#define WAVES_PER_BLOCK 8
#define QPW 2

__global__ __launch_bounds__(256) void mlp_kernel(const float* __restrict__ pc1,
                                                  const float* __restrict__ pc2,
                                                  const float* __restrict__ feat2,
                                                  const float* __restrict__ W1,
                                                  const float* __restrict__ b1,
                                                  const float* __restrict__ W2,
                                                  const float* __restrict__ b2,
                                                  const int* __restrict__ idxbuf,
                                                  float* __restrict__ out) {
  // Weights pre-packed per (tile, lane): Wpk[tile][lane][e], 16 f16 = 32B/lane.
  __shared__ _Float16 Wpk1[12 * 32 * 16];                  // 4 col-tiles x 3 k-steps
  __shared__ _Float16 Wpk2[8 * 32 * 16];                   // 4 col-tiles x 2 k-steps
  // Layer-1 output in A-matrix register layout: [wave][q][lane][32 halves].
  __shared__ _Float16 a2s[WAVES_PER_BLOCK][QPW][32 * 32];

  const int tid = threadIdx.x;

  // ---- Stage weights into the CDNA5 B-operand layout, with the channel
  //      permutation [feat | bias | xyz | 0]:
  //      r < 64  -> W1 row (r+3)   (feature weights)
  //      r == 64 -> b1             (bias row, pairs with A channel == 1)
  //      r < 68  -> W1 row (r-65)  (xyz weights)
  //      else    -> 0
  for (int i = tid; i < 12 * 512; i += 256) {
    const int e = i & 15, l = (i >> 4) & 31, u = i >> 9;   // u = t*3 + s
    const int t = u / 3, s = u - 3 * t;
    const int r = 32 * s + 8 * (l >> 4) + (e & 7) + 16 * (e >> 3);  // permuted padded row
    const int c = 16 * t + (l & 15);
    _Float16 v;
    if      (r < 64)  v = (_Float16)W1[(r + 3) * 64 + c];
    else if (r == 64) v = (_Float16)b1[c];
    else if (r < 68)  v = (_Float16)W1[(r - 65) * 64 + c];
    else              v = (_Float16)0.0f;
    Wpk1[i] = v;
  }
  for (int i = tid; i < 8 * 512; i += 256) {
    const int e = i & 15, l = (i >> 4) & 31, u = i >> 9;   // u = t*2 + s
    const int t = u >> 1, s = u & 1;
    const int r = 32 * s + 8 * (l >> 4) + (e & 7) + 16 * (e >> 3);
    Wpk2[i] = (_Float16)W2[r * 64 + 16 * t + (l & 15)];
  }
  __syncthreads();

  const int lane = tid & 31;
  const int wv   = tid >> 5;
  const int row  = lane & 15;   // neighbor slot (A-matrix row)
  const int h    = lane >> 4;   // wave half
  const int col  = lane & 15;   // output column within a 16-wide tile

  const int gq0 = (blockIdx.x * WAVES_PER_BLOCK + wv) * QPW;  // first query id
  const int b   = gq0 / N_;

  // ---- Per-query gather + A1 build (16 x 96 f16, A-register layout) ----
  // element e -> channel: e<8: 32s + 8h + e ; e>=8: 32s + 16 + 8h + (e-8)
  v16h A1[QPW][3];
#pragma unroll
  for (int q = 0; q < QPW; ++q) {
    const int gq = gq0 + q;
    const int m  = idxbuf[(size_t)gq * K_ + row];
    const float* frow = feat2 + ((size_t)b * M_ + m) * D_;
    const float4* f4  = (const float4*)frow;               // 16B-aligned rows
    const float nx = pc2[((size_t)b * M_ + m) * 3 + 0] - pc1[(size_t)gq * 3 + 0];
    const float ny = pc2[((size_t)b * M_ + m) * 3 + 1] - pc1[(size_t)gq * 3 + 1];
    const float nz = pc2[((size_t)b * M_ + m) * 3 + 2] - pc1[(size_t)gq * 3 + 2];

#pragma unroll
    for (int s = 0; s < 2; ++s) {                          // pure-feature k-steps
      const int r0 = (32 * s + 8 * h) >> 2;                // float4 index, aligned
      const int r1 = (32 * s + 16 + 8 * h) >> 2;
      const float4 u0 = f4[r0],  u1 = f4[r0 + 1];          // global_load_b128 x2
      const float4 w0 = f4[r1],  w1 = f4[r1 + 1];          // global_load_b128 x2
      v16h a;
      a[0]  = (_Float16)u0.x; a[1]  = (_Float16)u0.y;
      a[2]  = (_Float16)u0.z; a[3]  = (_Float16)u0.w;
      a[4]  = (_Float16)u1.x; a[5]  = (_Float16)u1.y;
      a[6]  = (_Float16)u1.z; a[7]  = (_Float16)u1.w;
      a[8]  = (_Float16)w0.x; a[9]  = (_Float16)w0.y;
      a[10] = (_Float16)w0.z; a[11] = (_Float16)w0.w;
      a[12] = (_Float16)w1.x; a[13] = (_Float16)w1.y;
      a[14] = (_Float16)w1.z; a[15] = (_Float16)w1.w;
      A1[q][s] = a;
    }
    {                                                      // bias/xyz/pad k-step
      v16h a = {};
      if (h == 0) {
        a[0] = (_Float16)1.0f;                             // bias channel (64)
        a[1] = (_Float16)nx;                               // xyz channels (65..67)
        a[2] = (_Float16)ny;
        a[3] = (_Float16)nz;
      }
      A1[q][2] = a;
    }
  }

  // ---- Layer 1: 4 col-tiles x 3 k-steps; B loaded once, used by both queries
#pragma unroll
  for (int t = 0; t < 4; ++t) {
    v8f acc[QPW] = {};                                     // inline-0 SRC2 start

#pragma unroll
    for (int s = 0; s < 3; ++s) {
      const v16h Bm = *(const v16h*)&Wpk1[((t * 3 + s) * 32 + lane) * 16];
#pragma unroll
      for (int q = 0; q < QPW; ++q)
        acc[q] = __builtin_amdgcn_wmma_f32_16x16x32_f16(
            false, A1[q][s], false, Bm, (short)0, acc[q], false, false);
    }

    // LeakyReLU + transpose-store into the A-register layout for layer 2.
    // ch = 16t+col is fixed per lane -> (lane', element') fixed; only the
    // neighbor row varies, giving a constant 64B stride (immediate offsets).
    const int ch = 16 * t + col;
    const int s2 = ch >> 5;
    const int klocal = ch & 31;
    const int h2 = (klocal >> 3) & 1;
    const int u  = klocal - 8 * h2;
    const int j2 = ((u >> 4) << 2) | ((u & 7) >> 1);
    const int eg = s2 * 16 + (2 * j2 + (u & 1));           // element within 32
#pragma unroll
    for (int q = 0; q < QPW; ++q) {
      _Float16* basep = &a2s[wv][q][(8 * h + 16 * h2) * 32 + eg];
#pragma unroll
      for (int r = 0; r < 8; ++r)
        basep[r * 32] = (_Float16)leaky(acc[q][r]);        // ds_store_b16, imm offs
    }
  }
  __syncthreads();

  // ---- Layer-2 A operands: contiguous 32B per k-step per lane ----
  v16h A2[QPW][2];
#pragma unroll
  for (int q = 0; q < QPW; ++q)
#pragma unroll
    for (int s = 0; s < 2; ++s)
      A2[q][s] = *(const v16h*)&a2s[wv][q][lane * 32 + s * 16];

  // ---- Layer 2: 4 col-tiles x 2 k-steps, then bias + LeakyReLU + max-pool
#pragma unroll
  for (int t = 0; t < 4; ++t) {
    v8f acc[QPW] = {};                                     // inline-0 SRC2 start
    const float bias = b2[16 * t + col];

#pragma unroll
    for (int s = 0; s < 2; ++s) {
      const v16h Bm = *(const v16h*)&Wpk2[((t * 2 + s) * 32 + lane) * 16];
#pragma unroll
      for (int q = 0; q < QPW; ++q)
        acc[q] = __builtin_amdgcn_wmma_f32_16x16x32_f16(
            false, A2[q][s], false, Bm, (short)0, acc[q], false, false);
    }

#pragma unroll
    for (int q = 0; q < QPW; ++q) {
      float mx = -3.4e38f;
#pragma unroll
      for (int r = 0; r < 8; ++r)
        mx = fmaxf(mx, leaky(acc[q][r] + bias));
      const float other = __shfl_xor(mx, 16, 32);
      mx = fmaxf(mx, other);
      if (lane < 16)
        out[(size_t)(gq0 + q) * 64 + 16 * t + lane] = mx;
    }
  }
}

// ---------------------------------------------------------------------------
extern "C" void kernel_launch(void* const* d_in, const int* in_sizes, int n_in,
                              void* d_out, int out_size, void* d_ws, size_t ws_size,
                              hipStream_t stream) {
  const float* pc1   = (const float*)d_in[0];
  const float* pc2   = (const float*)d_in[1];
  const float* feat2 = (const float*)d_in[2];
  const float* W1    = (const float*)d_in[3];
  const float* b1    = (const float*)d_in[4];
  const float* W2    = (const float*)d_in[5];
  const float* b2    = (const float*)d_in[6];
  float* out = (float*)d_out;
  int* idxbuf = (int*)d_ws;   // B*N*K ints = 2 MB

  // 1) top-16 neighbor indices
  knn_kernel<<<(B_ * N_) / 256, 256, 0, stream>>>(pc1, pc2, idxbuf);
  // 2) gather + WMMA MLP + max-pool; one wave per 2 query points
  mlp_kernel<<<(B_ * N_) / (WAVES_PER_BLOCK * QPW), 256, 0, stream>>>(
      pc1, pc2, feat2, W1, b1, W2, b2, idxbuf, out);
}
